// ProbSparseSelfAttention_7456063226393
// MI455X (gfx1250) — compile-verified, matrix operand order checked by reading, not run on hardware
//
#include <hip/hip_runtime.h>
#include <hip/hip_bf16.h>
#include <math.h>
#include <stdint.h>

// Problem constants (reference: B=4, S=2048, D=512, H=8, dk=64, u=floor(7*ln(2048))=53)
#define BB 4
#define SS 2048
#define DD 512
#define HH 8
#define DKX 64
#define UU 53

typedef __attribute__((ext_vector_type(16))) __bf16 bf16x16;
typedef __attribute__((ext_vector_type(8)))  __bf16 bf16x8;
typedef __attribute__((ext_vector_type(8)))  float  f32x8;
typedef __attribute__((ext_vector_type(4)))  int    i32x4;

// ---------------------------------------------------------------------------
// CDNA5 async global->LDS path (ASYNCcnt-tracked, no VGPR roundtrip).
// The builtin expects (v4i addrspace(1)*, v4i addrspace(3)*, i32, i32);
// build exact-typed pointers via integer round-trips (flat->LDS mapping is
// the low 32 bits of the address, per ISA 10.2, so the 32-bit truncation is correct).
// ---------------------------------------------------------------------------
#ifdef __has_builtin
#if __has_builtin(__builtin_amdgcn_global_load_async_to_lds_b128) && \
    __has_builtin(__builtin_amdgcn_s_wait_asynccnt)
#define USE_ASYNC_LDS 1
#endif
#endif

#ifdef USE_ASYNC_LDS
typedef __attribute__((address_space(1))) i32x4* gbl_v4i_ptr;
typedef __attribute__((address_space(3))) i32x4* lds_v4i_ptr;
#endif

static __device__ inline void copy16_to_lds(__bf16* dst_lds, const __bf16* src) {
#ifdef USE_ASYNC_LDS
  __builtin_amdgcn_global_load_async_to_lds_b128(
      (gbl_v4i_ptr)(uintptr_t)src,
      (lds_v4i_ptr)(uint32_t)(uintptr_t)dst_lds, 0, 0);
#else
  *(bf16x8*)dst_lds = *(const bf16x8*)src;   // global_load_b128 + ds_store_b128
#endif
}

static __device__ inline void async_copy_wait() {
#ifdef USE_ASYNC_LDS
  __builtin_amdgcn_s_wait_asynccnt(0);
#endif
}

// ---------------------------------------------------------------------------
// WMMA helpers (gfx1250: V_WMMA_F32_16X16X32_BF16, wave32)
// A-layout (ISA 7.12.2, 16-bit A 16x32): lanes 0-15 = row M=lane, K=0..7 & 16..23;
// lanes 16-31 = row M=lane-16, K=8..15 & 24..31.  B operand is fed as rows of
// B^T (N x K) using the mirrored layout.  Works for global or LDS pointers.
// ---------------------------------------------------------------------------
static __device__ inline bf16x16 load_tile16x32(const __bf16* __restrict__ p,
                                                int stride, int row0, int k0,
                                                int lane) {
  int r  = row0 + (lane & 15);
  int kh = (lane >> 4) << 3;                 // 0 or 8
  const __bf16* base = p + (size_t)r * stride + (k0 + kh);
  bf16x8 lo = *(const bf16x8*)(base);        // K = k0+kh .. +7   (b128 load)
  bf16x8 hi = *(const bf16x8*)(base + 16);   // K = k0+kh+16 .. +23
  return __builtin_shufflevector(lo, hi, 0,1,2,3,4,5,6,7,8,9,10,11,12,13,14,15);
}

static __device__ inline f32x8 wmma_bf16(bf16x16 a, bf16x16 b, f32x8 c) {
  // 8 args: (neg_a, A, neg_b, B, c_mod, C, reuse_a, reuse_b)
  return __builtin_amdgcn_wmma_f32_16x16x32_bf16(false, a, false, b,
                                                 (short)0, c, false, false);
}

// ---------------------------------------------------------------------------
// Generic bf16 GEMM: out = A[M,K] @ Bt[N,K]^T + bias[N]   (K <= 512)
// mode 0: f32 row-major [M,N];  mode 1: bf16 head-major [B,H,S,dk];
// mode 2: f32 head-major.
// 8 waves/block, all waves share one 16-row B tile staged in LDS.
// Grid must be exact: (M/16)*(N/16)/8 blocks, and (M/16) % 8 == 0.
// ---------------------------------------------------------------------------
__global__ __launch_bounds__(256)
void gemm_bf16(const __bf16* __restrict__ A, int lda,
               const __bf16* __restrict__ Bt, int ldb,
               const float* __restrict__ bias,
               float* __restrict__ outF, __bf16* __restrict__ outH,
               int M, int N, int K, int mode)
{
  __shared__ __attribute__((aligned(16))) __bf16 smB[16 * 512];

  int waveId = threadIdx.x >> 5;
  int lane   = threadIdx.x & 31;
  int mtiles = M >> 4;
  int tileBase = blockIdx.x * 8;
  int wid = tileBase + waveId;
  int tm = wid % mtiles, tn = wid / mtiles;      // tn uniform across block
  int m0 = tm << 4, n0 = tn << 4;
  int n0blk = (tileBase / mtiles) << 4;

  // Stage the 16 x K weight tile (shared by all 8 waves) into LDS.
  int vecsPerRow = K >> 3;
  for (int c = threadIdx.x; c < 16 * vecsPerRow; c += 256) {
    int row = c / vecsPerRow;
    int off = (c - row * vecsPerRow) << 3;
    copy16_to_lds(&smB[row * K + off], Bt + (size_t)(n0blk + row) * ldb + off);
  }
  async_copy_wait();
  __syncthreads();

  f32x8 c = {0.f,0.f,0.f,0.f,0.f,0.f,0.f,0.f};
  for (int k0 = 0; k0 < K; k0 += 32) {
    bf16x16 a = load_tile16x32(A,   lda, m0, k0, lane);   // global_load_b128
    bf16x16 b = load_tile16x32(smB, K,   0,  k0, lane);   // ds_load_b128
    if (k0 + 32 < K)  // global_prefetch_b8 for next A chunk
      __builtin_prefetch(A + (size_t)(m0 + (lane & 15)) * lda + k0 + 32, 0, 0);
    c = wmma_bf16(a, b, c);
  }

  // C layout: VGPR r, lanes 0-15 -> (M=r, N=lane); lanes 16-31 -> (M=8+r, N=lane-16)
  int col = n0 + (lane & 15);
  int mh  = (lane >> 4) << 3;
  float bv = bias ? bias[col] : 0.0f;
#pragma unroll
  for (int r = 0; r < 8; ++r) {
    int m = m0 + mh + r;
    float v = c[r] + bv;
    if (mode == 0) {
      outF[(size_t)m * N + col] = v;
    } else {
      int b_ = m >> 11, s = m & (SS - 1);
      int h  = col >> 6, d = col & (DKX - 1);
      size_t idx = (((size_t)(b_ * HH + h)) * SS + s) * DKX + d;
      if (mode == 1) outH[idx] = (__bf16)v;
      else           outF[idx] = v;
    }
  }
}

// ---------------------------------------------------------------------------
// Row-max of Q K^T / 8 per (b,h).  8 waves/block handle 8 adjacent query
// tiles; K-panel is streamed through LDS in double-buffered 128-row chunks
// (16 KB each), so global traffic for K drops 8x and the inner loop reads
// B operands with ds_load_b128.  Grid: BB*HH*16 blocks exactly.
// ---------------------------------------------------------------------------
__global__ __launch_bounds__(256)
void rowmax_k(const __bf16* __restrict__ Qh, const __bf16* __restrict__ Kh,
              float* __restrict__ rowmax)
{
  __shared__ __attribute__((aligned(16))) __bf16 smK[2][128 * DKX];

  int waveId = threadIdx.x >> 5;
  int lane   = threadIdx.x & 31;
  int blk = blockIdx.x;            // = bh*16 + blockTile
  int bh  = blk >> 4;
  int it  = (blk & 15) * 8 + waveId;
  const __bf16* Qb = Qh + (size_t)bh * SS * DKX;
  const __bf16* Kb = Kh + (size_t)bh * SS * DKX;

  bf16x16 a0 = load_tile16x32(Qb, DKX, it * 16, 0,  lane);
  bf16x16 a1 = load_tile16x32(Qb, DKX, it * 16, 32, lane);

  // cooperative chunk fill: 128 rows x 64 cols = 1024 b128 vectors / block
  auto fill_chunk = [&](__bf16* dst, int j0) {
    for (int cc = threadIdx.x; cc < 1024; cc += 256) {
      int row = cc >> 3;
      int off = (cc & 7) << 3;
      copy16_to_lds(dst + row * DKX + off, Kb + (size_t)(j0 + row) * DKX + off);
    }
  };

  fill_chunk(smK[0], 0);

  f32x8 vmax = {-1e30f,-1e30f,-1e30f,-1e30f,-1e30f,-1e30f,-1e30f,-1e30f};
  const int nChunks = SS / 128;            // 16
  for (int ch = 0; ch < nChunks; ++ch) {
    async_copy_wait();                     // our wave's fills have landed
    __syncthreads();                       // everyone's fills visible
    if (ch + 1 < nChunks) fill_chunk(smK[(ch + 1) & 1], (ch + 1) * 128);
    const __bf16* kb = smK[ch & 1];
#pragma unroll
    for (int jt = 0; jt < 8; ++jt) {
      bf16x16 b0 = load_tile16x32(kb, DKX, jt * 16, 0,  lane);  // ds_load_b128
      bf16x16 b1 = load_tile16x32(kb, DKX, jt * 16, 32, lane);
      f32x8 c = {0.f,0.f,0.f,0.f,0.f,0.f,0.f,0.f};
      c = wmma_bf16(a0, b0, c);
      c = wmma_bf16(a1, b1, c);
#pragma unroll
      for (int r = 0; r < 8; ++r) vmax[r] = fmaxf(vmax[r], c[r]);
    }
  }

#pragma unroll
  for (int mask = 1; mask <= 8; mask <<= 1)
#pragma unroll
    for (int r = 0; r < 8; ++r)
      vmax[r] = fmaxf(vmax[r], __shfl_xor(vmax[r], mask, 32));
  if ((lane & 15) == 0) {
    int rbase = it * 16 + ((lane >> 4) << 3);
#pragma unroll
    for (int r = 0; r < 8; ++r)
      rowmax[(size_t)bh * SS + rbase + r] = vmax[r] * 0.125f;  // /sqrt(dk)
  }
}

// ---------------------------------------------------------------------------
// Column sums: Ksum[bh][d] (from bf16), sumV[bh][d] (from f32)
// ---------------------------------------------------------------------------
__global__ __launch_bounds__(256)
void colsum_bf16(const __bf16* __restrict__ X, float* __restrict__ out) {
  __shared__ float sm[256];
  int bh = blockIdx.x;
  int d = threadIdx.x & 63, part = threadIdx.x >> 6;
  const __bf16* p = X + (size_t)bh * SS * DKX + (size_t)part * (SS / 4) * DKX + d;
  float s = 0.f;
  for (int i = 0; i < SS / 4; ++i) s += (float)p[(size_t)i * DKX];
  sm[threadIdx.x] = s;
  __syncthreads();
  if (part == 0) out[(size_t)bh * DKX + d] = sm[d] + sm[64 + d] + sm[128 + d] + sm[192 + d];
}

__global__ __launch_bounds__(256)
void colsum_f32(const float* __restrict__ X, float* __restrict__ out) {
  __shared__ float sm[256];
  int bh = blockIdx.x;
  int d = threadIdx.x & 63, part = threadIdx.x >> 6;
  const float* p = X + (size_t)bh * SS * DKX + (size_t)part * (SS / 4) * DKX + d;
  float s = 0.f;
  for (int i = 0; i < SS / 4; ++i) s += p[(size_t)i * DKX];
  sm[threadIdx.x] = s;
  __syncthreads();
  if (part == 0) out[(size_t)bh * DKX + d] = sm[d] + sm[64 + d] + sm[128 + d] + sm[192 + d];
}

// ---------------------------------------------------------------------------
// sparsity[i] = mean_j(scores) - max_j(scores)/ln(S);  mean via Q_i . Ksum
// ---------------------------------------------------------------------------
__global__ __launch_bounds__(256)
void sparsity_k(const __bf16* __restrict__ Qh, const float* __restrict__ Ksum,
                const float* __restrict__ rowmax, float* __restrict__ sp)
{
  int idx = blockIdx.x * blockDim.x + threadIdx.x;   // bh*SS + i
  if (idx >= BB * HH * SS) return;
  int bh = idx >> 11;
  const __bf16* q = Qh + (size_t)idx * DKX;
  const float* ks = Ksum + (size_t)bh * DKX;
  float acc = 0.f;
#pragma unroll 8
  for (int d = 0; d < DKX; ++d) acc += (float)q[d] * ks[d];
  const float inv_log_s = 1.0f / 7.6246189861593985f;  // ln(2048)
  sp[idx] = acc * (1.0f / (8.0f * (float)SS)) - rowmax[idx] * inv_log_s;
}

// ---------------------------------------------------------------------------
// Top-53 per (b,h), descending, low-index tie-break (matches jax.lax.top_k).
// One wave per (b,h); iterative selection over lane-strided private values.
// ---------------------------------------------------------------------------
__global__ __launch_bounds__(32)
void topk_k(const float* __restrict__ sp, int* __restrict__ topidx)
{
  int bh = blockIdx.x;
  int lane = threadIdx.x;
  float vals[SS / 32];
  for (int t = 0; t < SS / 32; ++t) vals[t] = sp[(size_t)bh * SS + t * 32 + lane];
  for (int k = 0; k < UU; ++k) {
    float best = -3e30f; int bidx = 0x7fffffff;
    for (int t = 0; t < SS / 32; ++t) {
      float v = vals[t]; int i = t * 32 + lane;
      if (v > best || (v == best && i < bidx)) { best = v; bidx = i; }
    }
#pragma unroll
    for (int mask = 16; mask >= 1; mask >>= 1) {
      float ov = __shfl_xor(best, mask, 32);
      int   oi = __shfl_xor(bidx, mask, 32);
      if (ov > best || (ov == best && oi < bidx)) { best = ov; bidx = oi; }
    }
    if (lane == 0) topidx[bh * UU + k] = bidx;
    if ((bidx & 31) == lane) vals[bidx >> 5] = -3e30f;   // remove winner
  }
}

// ---------------------------------------------------------------------------
// ctx default: every row gets uniform-softmax result sumV/S  (bf16 [B,S,D])
// ---------------------------------------------------------------------------
__global__ __launch_bounds__(256)
void fill_ctx(const float* __restrict__ sumV, __bf16* __restrict__ ctx)
{
  int idx = blockIdx.x * blockDim.x + threadIdx.x;
  if (idx >= BB * SS * DD) return;
  int dcol = idx & (DD - 1);
  int b_ = idx >> 20;               // / (S*D)
  int h = dcol >> 6, d = dcol & 63;
  ctx[idx] = (__bf16)(sumV[(size_t)(b_ * HH + h) * DKX + d] * (1.0f / (float)SS));
}

// ---------------------------------------------------------------------------
// Special rows i<u: z = Q_i.K_j/8, out = (sumV + (e^z-1) V_j) / (e^z + S-1)
// One wave per (b,h,i); full-wave shfl reduction for the 64-dim dot.
// ---------------------------------------------------------------------------
__global__ __launch_bounds__(256)
void special_rows(const __bf16* __restrict__ Qh, const __bf16* __restrict__ Kh,
                  const float* __restrict__ Vh, const float* __restrict__ sumV,
                  const int* __restrict__ topidx, __bf16* __restrict__ ctx)
{
  int wid = blockIdx.x * (blockDim.x >> 5) + (threadIdx.x >> 5);
  if (wid >= BB * HH * UU) return;
  int lane = threadIdx.x & 31;
  int bh = wid / UU;
  int i  = wid - bh * UU;
  int j  = topidx[wid];
  int b_ = bh >> 3, h = bh & 7;
  const __bf16* q  = Qh + ((size_t)bh * SS + i) * DKX;
  const __bf16* kk = Kh + ((size_t)bh * SS + j) * DKX;
  float part = (float)q[lane * 2]     * (float)kk[lane * 2]
             + (float)q[lane * 2 + 1] * (float)kk[lane * 2 + 1];
#pragma unroll
  for (int mask = 16; mask >= 1; mask >>= 1) part += __shfl_xor(part, mask, 32);
  float z = part * 0.125f;
  float e = expf(z);
  float inv = 1.0f / (e + (float)(SS - 1));
  const float* vrow = Vh + ((size_t)bh * SS + j) * DKX;
  const float* sv = sumV + (size_t)bh * DKX;
#pragma unroll
  for (int t = 0; t < 2; ++t) {
    int d = lane * 2 + t;
    float val = (sv[d] + (e - 1.0f) * vrow[d]) * inv;
    ctx[((size_t)b_ * SS + i) * DD + h * DKX + d] = (__bf16)val;
  }
}

// ---------------------------------------------------------------------------
__global__ __launch_bounds__(256)
void f2bf(const float* __restrict__ in, __bf16* __restrict__ out, int n)
{
  int i = blockIdx.x * blockDim.x + threadIdx.x;
  if (i < n) out[i] = (__bf16)in[i];
}

// ---------------------------------------------------------------------------
extern "C" void kernel_launch(void* const* d_in, const int* in_sizes, int n_in,
                              void* d_out, int out_size, void* d_ws, size_t ws_size,
                              hipStream_t stream)
{
  const float* x  = (const float*)d_in[0];
  const float* wq = (const float*)d_in[1];
  const float* bq = (const float*)d_in[2];
  const float* wk = (const float*)d_in[3];
  const float* bk = (const float*)d_in[4];
  const float* wv = (const float*)d_in[5];
  const float* bv = (const float*)d_in[6];
  const float* wo = (const float*)d_in[7];
  const float* bo = (const float*)d_in[8];
  float* out = (float*)d_out;

  char* ws = (char*)d_ws;
  size_t off = 0;
  auto alloc = [&](size_t bytes) -> char* {
    char* p = ws + off;
    off = (off + bytes + 255) & ~(size_t)255;
    return p;
  };

  const int NX = BB * SS * DD;          // 4,194,304
  const int NW = DD * DD;               // 262,144
  __bf16* xb   = (__bf16*)alloc((size_t)NX * 2);
  __bf16* wqb  = (__bf16*)alloc((size_t)NW * 2);
  __bf16* wkb  = (__bf16*)alloc((size_t)NW * 2);
  __bf16* wvb  = (__bf16*)alloc((size_t)NW * 2);
  __bf16* wob  = (__bf16*)alloc((size_t)NW * 2);
  __bf16* Qh   = (__bf16*)alloc((size_t)NX * 2);
  __bf16* Kh   = (__bf16*)alloc((size_t)NX * 2);
  float*  Vh   = (float*) alloc((size_t)NX * 4);
  float*  Ksum = (float*) alloc((size_t)BB * HH * DKX * 4);
  float*  sumV = (float*) alloc((size_t)BB * HH * DKX * 4);
  float*  rmax = (float*) alloc((size_t)BB * HH * SS * 4);
  float*  sp   = (float*) alloc((size_t)BB * HH * SS * 4);
  int*    tidx = (int*)   alloc((size_t)BB * HH * UU * 4);
  __bf16* ctx  = (__bf16*)alloc((size_t)NX * 2);
  (void)ws_size; (void)in_sizes; (void)n_in; (void)out_size;

  // 1) f32 -> bf16 conversions
  f2bf<<<(NX + 255) / 256, 256, 0, stream>>>(x,  xb,  NX);
  f2bf<<<(NW + 255) / 256, 256, 0, stream>>>(wq, wqb, NW);
  f2bf<<<(NW + 255) / 256, 256, 0, stream>>>(wk, wkb, NW);
  f2bf<<<(NW + 255) / 256, 256, 0, stream>>>(wv, wvb, NW);
  f2bf<<<(NW + 255) / 256, 256, 0, stream>>>(wo, wob, NW);

  // 2) QKV projections (WMMA).  M=8192 N=512 K=512 -> 16384 tiles, 8 waves/blk
  const int M = BB * SS, N = DD, K = DD;
  const int gemmBlocks = (M / 16) * (N / 16) / 8;   // exact, (M/16)%8==0
  gemm_bf16<<<gemmBlocks, 256, 0, stream>>>(xb, DD, wqb, DD, bq, nullptr, Qh, M, N, K, 1);
  gemm_bf16<<<gemmBlocks, 256, 0, stream>>>(xb, DD, wkb, DD, bk, nullptr, Kh, M, N, K, 1);
  gemm_bf16<<<gemmBlocks, 256, 0, stream>>>(xb, DD, wvb, DD, bv, Vh, nullptr, M, N, K, 2);

  // 3) column sums
  colsum_bf16<<<BB * HH, 256, 0, stream>>>(Kh, Ksum);
  colsum_f32 <<<BB * HH, 256, 0, stream>>>(Vh, sumV);

  // 4) row-max of scores (WMMA, LDS-staged K panel): 512 blocks exactly
  rowmax_k<<<BB * HH * (SS / 128), 256, 0, stream>>>(Qh, Kh, rmax);

  // 5) sparsity + top-k
  sparsity_k<<<(BB * HH * SS + 255) / 256, 256, 0, stream>>>(Qh, Ksum, rmax, sp);
  topk_k<<<BB * HH, 32, 0, stream>>>(sp, tidx);

  // 6) context: uniform rows, then the 53 masked rows per (b,h)
  fill_ctx<<<(NX + 255) / 256, 256, 0, stream>>>(sumV, ctx);
  special_rows<<<(BB * HH * UU + 7) / 8, 256, 0, stream>>>(Qh, Kh, Vh, sumV, tidx, ctx);

  // 7) output projection (WMMA) straight to d_out (f32 row-major [B,S,D])
  gemm_bf16<<<gemmBlocks, 256, 0, stream>>>(ctx, DD, wob, DD, bo, out, nullptr, M, N, K, 0);
}
